// WindowAttentionGhost_64467459113258
// MI455X (gfx1250) — compile-verified, matrix operand order checked by reading, not run on hardware
//
#include <hip/hip_runtime.h>

typedef _Float16 h16;
typedef __attribute__((ext_vector_type(16))) _Float16 v16h;
typedef __attribute__((ext_vector_type(8)))  _Float16 v8h;
typedef __attribute__((ext_vector_type(8)))  float    v8f;

// ---------------------------------------------------------------------------
// WMMA fragment loader.
// 16-bit A/B operand layout (CDNA5 ISA 7.12.2): lane r (0..15) holds row r with
// halves K = {k0..k0+7} in v[0..3] and K = {k0+16..k0+23} in v[4..7];
// lane r+16 holds the same row with an extra +8 offset on both chunks.
// "row" is the M-row for A, or the N-column (stored as a row of K values) for
// B — both load identically from row-major [row][K] storage.
// ---------------------------------------------------------------------------
__device__ __forceinline__ v16h frag_row(const h16* __restrict__ row, int k0, int off8) {
    v8h lo = *(const v8h*)(row + k0 + off8);
    v8h hi = *(const v8h*)(row + k0 + 16 + off8);
    v16h f;
#pragma unroll
    for (int j = 0; j < 8; ++j) { f[j] = lo[j]; f[8 + j] = hi[j]; }
    return f;
}

// ============================ prep kernels =================================

// Convert Wq|Wk|Wv (stacked 576x384) and proj_w (384x384) to f16, row-major.
__global__ void wag_prep_w(const float* __restrict__ Wq, const float* __restrict__ Wk,
                           const float* __restrict__ Wv, const float* __restrict__ Pw,
                           h16* __restrict__ Wqkv, h16* __restrict__ Ph) {
    int idx = blockIdx.x * 256 + threadIdx.x;
    if (idx < 576 * 384) {
        int c = idx / 384, k = idx % 384;
        float v = (c < 192) ? Wq[c * 384 + k]
                : (c < 384) ? Wk[(c - 192) * 384 + k]
                            : Wv[(c - 384) * 384 + k];
        Wqkv[idx] = (h16)v;
    } else if (idx < 576 * 384 + 384 * 384) {
        int j = idx - 576 * 384;
        Ph[j] = (h16)Pw[j];
    }
}

// CPB MLP: tbl[t][h] = 16*sigmoid( relu(coords@W1^T + b1) @ W2^T )
__global__ void wag_cpb(const float* __restrict__ ct, const float* __restrict__ w1,
                        const float* __restrict__ b1, const float* __restrict__ w2,
                        float* __restrict__ tbl) {
    int idx = blockIdx.x * 256 + threadIdx.x;
    if (idx >= 169 * 12) return;
    int t = idx / 12, h = idx % 12;
    float c0 = ct[t * 2], c1 = ct[t * 2 + 1];
    float acc = 0.f;
    for (int j = 0; j < 512; ++j) {
        float hv = fmaxf(c0 * w1[j * 2] + c1 * w1[j * 2 + 1] + b1[j], 0.f);
        acc += hv * w2[h * 512 + j];
    }
    tbl[idx] = 16.f / (1.f + __expf(-acc));
}

// Scatter table -> rpb[h][i][j]; also per-head logit scale.
__global__ void wag_rpb(const float* __restrict__ tbl, const int* __restrict__ relidx,
                        const float* __restrict__ ls, float* __restrict__ rpb,
                        float* __restrict__ scale) {
    int idx = blockIdx.x * 256 + threadIdx.x;
    if (idx < 12) scale[idx] = __expf(fminf(ls[idx], 4.6051702f)); // ln(100)
    if (idx >= 12 * 2401) return;
    int h = idx / 2401, ij = idx % 2401;
    rpb[idx] = tbl[relidx[ij] * 12 + h];
}

// ============================ QKV GEMM + ghost ==============================
// grid (3136), block 128 (4 waves).  Each block: M=64 rows x full N=576.
// x tile staged in LDS ONCE (f16, 64x384) -> x read exactly once from HBM.
// W tiles (192x32) staged cooperatively in LDS per k-step; B frags via ds_load.
// Epilogue scatters into q/k/v f16 buffers [B][12][49][32] with the ghost
// duplicate: head h0 = val, head h0+6 = val*g[c].
__global__ __launch_bounds__(128) void wag_qkv(
    const float* __restrict__ x, const h16* __restrict__ Wh,
    const float* __restrict__ gq, const float* __restrict__ gk, const float* __restrict__ gv,
    h16* __restrict__ qh, h16* __restrict__ kh, h16* __restrict__ vh) {
    __shared__ __align__(16) h16 xs[64 * 392];   // stride 392 halves (16B aligned rows)
    __shared__ __align__(16) h16 wt[192 * 40];
    const int mblk = blockIdx.x;
    const int tid = threadIdx.x;
    const int w = tid >> 5, lane = tid & 31, r16 = lane & 15, off = (lane >> 4) * 8;
    const long rowbase = (long)mblk * 64;

    // stage full 64x384 x tile as f16 (6144 float4 chunks / 128 threads)
#pragma unroll
    for (int j = 0; j < 48; ++j) {
        int idx = j * 128 + tid;              // 0..6143
        int r = idx / 96, c4 = (idx % 96) * 4;
        const float4 vv = *(const float4*)(x + (rowbase + r) * 384 + c4);
        h16* d = &xs[r * 392 + c4];
        d[0] = (h16)vv.x; d[1] = (h16)vv.y; d[2] = (h16)vv.z; d[3] = (h16)vv.w;
    }

    int bb[8], nn[8];
#pragma unroll
    for (int i = 0; i < 8; ++i) {
        int gm = (int)rowbase + w * 16 + (lane >> 4) * 8 + i;
        bb[i] = gm / 49; nn[i] = gm % 49;
    }

    for (int g = 0; g < 3; ++g) {
        v8f acc[12] = {};
        for (int kt = 0; kt < 12; ++kt) {
            __syncthreads();                  // wt free to overwrite (also fences xs stores)
            // stage 192x32 W tile (768 v8h chunks / 128 threads)
#pragma unroll
            for (int j = 0; j < 6; ++j) {
                int idx = j * 128 + tid;      // 0..767
                int row = idx >> 2, c8 = (idx & 3) * 8;
                *(v8h*)(&wt[row * 40 + c8]) =
                    *(const v8h*)(Wh + (long)(g * 192 + row) * 384 + kt * 32 + c8);
            }
            __syncthreads();
            v16h a = frag_row(&xs[(w * 16 + r16) * 392], kt * 32, off);
#pragma unroll
            for (int nt = 0; nt < 12; ++nt) {
                v16h b = frag_row(&wt[(nt * 16 + r16) * 40], 0, off);
                acc[nt] = __builtin_amdgcn_wmma_f32_16x16x32_f16(false, a, false, b,
                                                                 (short)0, acc[nt], false, false);
            }
        }
        // epilogue for this 192-column group (registers only; no LDS reads)
#pragma unroll
        for (int nt = 0; nt < 12; ++nt) {
            int gc = g * 192 + nt * 16 + r16;
            int mid = gc / 192, c = gc % 192;
            int h0 = c >> 5, d = c & 31;
            const float gg = (mid == 0 ? gq : mid == 1 ? gk : gv)[c];
            h16* dst = (mid == 0 ? qh : mid == 1 ? kh : vh);
#pragma unroll
            for (int i = 0; i < 8; ++i) {
                float val = acc[nt][i];
                long idx0 = (((long)bb[i] * 12 + h0) * 49 + nn[i]) * 32 + d;
                dst[idx0] = (h16)val;
                dst[idx0 + 6 * 49 * 32] = (h16)(val * gg);
            }
        }
    }
}

// ============================ attention =====================================
// grid (4096, 12), block 128 (4 waves).  One (window, head) per block.
__global__ __launch_bounds__(128) void wag_attn(
    const h16* __restrict__ qh, const h16* __restrict__ kh, const h16* __restrict__ vh,
    const float* __restrict__ rpb, const float* __restrict__ mask,
    const float* __restrict__ scale, h16* __restrict__ yh) {
    __shared__ __align__(16) h16 Qs[64 * 40];
    __shared__ __align__(16) h16 Ks[64 * 40];
    __shared__ __align__(16) h16 Vt[32 * 72];
    __shared__ __align__(16) float Sb[64 * 68];
    __shared__ __align__(16) h16 Ps[64 * 72];
    __shared__ float invq[64], invk[64];

    const int b = blockIdx.x, h = blockIdx.y;
    const int tid = threadIdx.x;
    const long hb = ((long)b * 12 + h) * (49 * 32);

    // load Q/K (rows 49..63 zero-padded)
#pragma unroll
    for (int j = 0; j < 2; ++j) {
        int idx = j * 128 + tid;              // 0..255
        int r = idx >> 2, c8 = (idx & 3) * 8;
        v8h z = {};
        v8h q = (r < 49) ? *(const v8h*)(qh + hb + r * 32 + c8) : z;
        v8h k = (r < 49) ? *(const v8h*)(kh + hb + r * 32 + c8) : z;
        *(v8h*)(&Qs[r * 40 + c8]) = q;
        *(v8h*)(&Ks[r * 40 + c8]) = k;
    }
    // V transposed: Vt[d][m] (cols 49..63 zero)
#pragma unroll
    for (int j = 0; j < 16; ++j) {
        int idx = j * 128 + tid;              // 0..2047
        int d = idx & 31, m = idx >> 5;
        float val = (m < 49) ? (float)vh[hb + (long)m * 32 + d] : 0.f;
        Vt[d * 72 + m] = (h16)val;
    }
    __syncthreads();

    // cosine-attention row norms; logit scale folded into q
    if (tid < 64) {
        float sq = 0.f, sk = 0.f;
        if (tid < 49) {
            for (int d = 0; d < 32; ++d) {
                float a = (float)Qs[tid * 40 + d]; sq += a * a;
                float c = (float)Ks[tid * 40 + d]; sk += c * c;
            }
        }
        invq[tid] = (tid < 49) ? scale[h] / sqrtf(fmaxf(sq, 1e-24f)) : 0.f;
        invk[tid] = (tid < 49) ? 1.f / sqrtf(fmaxf(sk, 1e-24f)) : 0.f;
    }
    __syncthreads();
#pragma unroll
    for (int j = 0; j < 16; ++j) {
        int idx = j * 128 + tid;
        int r = idx >> 5, d = idx & 31;
        Qs[r * 40 + d] = (h16)((float)Qs[r * 40 + d] * invq[r]);
        Ks[r * 40 + d] = (h16)((float)Ks[r * 40 + d] * invk[r]);
    }
    __syncthreads();

    const int w = tid >> 5, lane = tid & 31, r16 = lane & 15, off = (lane >> 4) * 8;

    // S = Q̂ @ K̂^T  (K-dim == 32 == head dim, one WMMA per 16x16 tile)
    v16h a = frag_row(&Qs[(w * 16 + r16) * 40], 0, off);
    v8f s[4];
#pragma unroll
    for (int nt = 0; nt < 4; ++nt) {
        v16h bf = frag_row(&Ks[(nt * 16 + r16) * 40], 0, off);
        v8f z = {};
        s[nt] = __builtin_amdgcn_wmma_f32_16x16x32_f16(false, a, false, bf,
                                                       (short)0, z, false, false);
    }
    // dump with rpb + window mask; pad with -1e30
    const int win = b & 63;
#pragma unroll
    for (int nt = 0; nt < 4; ++nt) {
        int n = nt * 16 + r16;
#pragma unroll
        for (int i = 0; i < 8; ++i) {
            int m = w * 16 + (lane >> 4) * 8 + i;
            float v = -1e30f;
            if (m < 49 && n < 49)
                v = s[nt][i] + rpb[((long)h * 49 + m) * 49 + n]
                             + mask[((long)win * 49 + m) * 49 + n];
            Sb[m * 68 + n] = v;
        }
    }
    __syncthreads();

    // row softmax -> f16 P
    if (tid < 64) {
        float mx = -3e30f;
        for (int n = 0; n < 64; ++n) mx = fmaxf(mx, Sb[tid * 68 + n]);
        float sum = 0.f;
        for (int n = 0; n < 64; ++n) {
            float e = __expf(Sb[tid * 68 + n] - mx);
            Sb[tid * 68 + n] = e; sum += e;
        }
        float r = 1.f / sum;
        for (int n = 0; n < 64; ++n) Ps[tid * 72 + n] = (h16)(Sb[tid * 68 + n] * r);
    }
    __syncthreads();

    // O = P @ V   (K = 64 padded -> two 32-chunks; N = 32 -> two n-tiles)
    v8f o[2] = {};
#pragma unroll
    for (int kt = 0; kt < 2; ++kt) {
        v16h ap = frag_row(&Ps[(w * 16 + r16) * 72], kt * 32, off);
#pragma unroll
        for (int nt = 0; nt < 2; ++nt) {
            v16h bv = frag_row(&Vt[(nt * 16 + r16) * 72], kt * 32, off);
            o[nt] = __builtin_amdgcn_wmma_f32_16x16x32_f16(false, ap, false, bv,
                                                           (short)0, o[nt], false, false);
        }
    }
#pragma unroll
    for (int nt = 0; nt < 2; ++nt) {
        int d = nt * 16 + r16;
#pragma unroll
        for (int i = 0; i < 8; ++i) {
            int m = w * 16 + (lane >> 4) * 8 + i;
            if (m < 49)
                yh[((long)b * 49 + m) * 384 + h * 32 + d] = (h16)o[nt][i];
        }
    }
}

// ============================ output projection =============================
// grid (3136), block 128.  M=64 x full N=384 per block; y staged once in LDS,
// proj_w tiles staged per k-step.  f32 out + bias.
__global__ __launch_bounds__(128) void wag_proj(
    const h16* __restrict__ yh, const h16* __restrict__ Ph,
    const float* __restrict__ pb, float* __restrict__ out) {
    __shared__ __align__(16) h16 ys[64 * 392];
    __shared__ __align__(16) h16 pt[192 * 40];
    const int mblk = blockIdx.x;
    const int tid = threadIdx.x;
    const int w = tid >> 5, lane = tid & 31, r16 = lane & 15, off = (lane >> 4) * 8;
    const long rowbase = (long)mblk * 64;

    // stage full 64x384 f16 y tile (3072 v8h chunks / 128 threads)
#pragma unroll
    for (int j = 0; j < 24; ++j) {
        int idx = j * 128 + tid;              // 0..3071
        int r = idx / 48, c8 = (idx % 48) * 8;
        *(v8h*)(&ys[r * 392 + c8]) = *(const v8h*)(yh + (rowbase + r) * 384 + c8);
    }

    for (int g = 0; g < 2; ++g) {
        v8f acc[12] = {};
        for (int kt = 0; kt < 12; ++kt) {
            __syncthreads();
#pragma unroll
            for (int j = 0; j < 6; ++j) {
                int idx = j * 128 + tid;      // 0..767
                int row = idx >> 2, c8 = (idx & 3) * 8;
                *(v8h*)(&pt[row * 40 + c8]) =
                    *(const v8h*)(Ph + (long)(g * 192 + row) * 384 + kt * 32 + c8);
            }
            __syncthreads();
            v16h a = frag_row(&ys[(w * 16 + r16) * 392], kt * 32, off);
#pragma unroll
            for (int nt = 0; nt < 12; ++nt) {
                v16h b = frag_row(&pt[(nt * 16 + r16) * 40], 0, off);
                acc[nt] = __builtin_amdgcn_wmma_f32_16x16x32_f16(false, a, false, b,
                                                                 (short)0, acc[nt], false, false);
            }
        }
#pragma unroll
        for (int nt = 0; nt < 12; ++nt) {
            int gc = g * 192 + nt * 16 + r16;
            float bias = pb[gc];
#pragma unroll
            for (int i = 0; i < 8; ++i) {
                long gm = rowbase + w * 16 + (lane >> 4) * 8 + i;
                out[gm * 384 + gc] = acc[nt][i] + bias;
            }
        }
    }
}

// ============================ launch ========================================
extern "C" void kernel_launch(void* const* d_in, const int* in_sizes, int n_in,
                              void* d_out, int out_size, void* d_ws, size_t ws_size,
                              hipStream_t stream) {
    const float* x      = (const float*)d_in[0];
    const float* mask   = (const float*)d_in[1];
    const float* Wq     = (const float*)d_in[2];
    const float* Wk     = (const float*)d_in[3];
    const float* Wv     = (const float*)d_in[4];
    const float* gq     = (const float*)d_in[5];
    const float* gk     = (const float*)d_in[6];
    const float* gv     = (const float*)d_in[7];
    const float* ls     = (const float*)d_in[8];
    const float* ct     = (const float*)d_in[9];
    const float* w1     = (const float*)d_in[10];
    const float* b1     = (const float*)d_in[11];
    const float* w2     = (const float*)d_in[12];
    const float* pw     = (const float*)d_in[13];
    const float* pb     = (const float*)d_in[14];
    const int*   relidx = (const int*)d_in[15];
    float* out = (float*)d_out;

    char* ws = (char*)d_ws;
    size_t o = 0;
    auto alloc = [&](size_t bytes) { size_t p = o; o += (bytes + 255) & ~(size_t)255; return p; };
    const size_t headElems = (size_t)4096 * 12 * 49 * 32; // 77,070,336

    h16*   Wqkv  = (h16*)(ws + alloc((size_t)576 * 384 * 2));
    h16*   Ph    = (h16*)(ws + alloc((size_t)384 * 384 * 2));
    float* tbl   = (float*)(ws + alloc((size_t)169 * 12 * 4));
    float* rpb   = (float*)(ws + alloc((size_t)12 * 49 * 49 * 4));
    float* scale = (float*)(ws + alloc((size_t)12 * 4));
    h16*   qh    = (h16*)(ws + alloc(headElems * 2));
    h16*   kh    = (h16*)(ws + alloc(headElems * 2));
    h16*   vh    = (h16*)(ws + alloc(headElems * 2));
    h16*   yh    = (h16*)(ws + alloc(headElems * 2));
    (void)in_sizes; (void)n_in; (void)out_size; (void)ws_size;

    wag_prep_w<<<1440, 256, 0, stream>>>(Wq, Wk, Wv, pw, Wqkv, Ph);
    wag_cpb<<<(169 * 12 + 255) / 256, 256, 0, stream>>>(ct, w1, b1, w2, tbl);
    wag_rpb<<<(12 * 2401 + 255) / 256, 256, 0, stream>>>(tbl, relidx, ls, rpb, scale);
    wag_qkv<<<3136, 128, 0, stream>>>(x, Wqkv, gq, gk, gv, qh, kh, vh);
    wag_attn<<<dim3(4096, 12), 128, 0, stream>>>(qh, kh, vh, rpb, mask, scale, yh);
    wag_proj<<<3136, 128, 0, stream>>>(yh, Ph, pb, out);
}
